// IDMRFLoss_738734375648
// MI455X (gfx1250) — matrix operand order repaired and checked by reading, not compiled, not run on hardware
//
#include <hip/hip_runtime.h>
#include <math.h>

// ---------------------------------------------------------------------------
// Types for CDNA5 WMMA
// ---------------------------------------------------------------------------
typedef __attribute__((ext_vector_type(16))) __bf16 v16bf;
typedef __attribute__((ext_vector_type(8)))  float  v8f;
typedef __attribute__((ext_vector_type(4)))  int    v4i_t;

#define INV_SIGMA (1.0f / (0.5f + 1e-5f))   // 1/(SIGMA+eps); BIAS == 1.0

// --- CDNA5 async global->LDS path (guarded; fallback = sync staging) -------
// Probe-learned signature: (int4 addrspace(1)*, int4 addrspace(3)*, i32, i32)
#if __has_builtin(__builtin_amdgcn_global_load_async_to_lds_b128)
#define HAVE_ASYNC_LDS 1
#define GASYNC_B128(gptr, lptr)                                          \
  __builtin_amdgcn_global_load_async_to_lds_b128(                        \
      (__attribute__((address_space(1))) v4i_t*)(gptr),                  \
      (__attribute__((address_space(3))) v4i_t*)(lptr), 0, 0)
#else
#define HAVE_ASYNC_LDS 0
#endif

#if __has_builtin(__builtin_amdgcn_s_wait_asynccnt)
#define WAIT_ASYNC(n) __builtin_amdgcn_s_wait_asynccnt(n)
#else
#define WAIT_ASYNC(n) asm volatile("s_wait_asynccnt %0" ::"n"(n) : "memory")
#endif

// float -> bf16 bits, round-to-nearest-even
__device__ __forceinline__ unsigned short f2bf(float f) {
  unsigned u = __float_as_uint(f);
  u += 0x7FFFu + ((u >> 16) & 1u);
  return (unsigned short)(u >> 16);
}

// order-preserving float<->uint keys (for atomicMin over possibly-signed floats)
__device__ __forceinline__ unsigned fkeyEnc(float f) {
  unsigned u = __float_as_uint(f);
  return (u & 0x80000000u) ? ~u : (u | 0x80000000u);
}
__device__ __forceinline__ float fkeyDec(unsigned k) {
  return (k & 0x80000000u) ? __uint_as_float(k & 0x7FFFFFFFu)
                           : __uint_as_float(~k);
}

__device__ __forceinline__ float waveAllSum(float v) {
#pragma unroll
  for (int off = 16; off; off >>= 1) v += __shfl_xor(v, off, 32);
  return v;
}

// ---------------------------------------------------------------------------
// Kernel 1: per-position channel normalization (layout [B,C,N] -> bf16 [B,N,C])
//   meanT over channels of tar; g=gen-meanT, t=tar-meanT; L2-normalize over C.
// One wave per spatial position; lanes stride channels.
// ---------------------------------------------------------------------------
template <int CC>
__global__ __launch_bounds__(256)
void mrf_normalize(const float* __restrict__ gen, const float* __restrict__ tar,
                   unsigned short* __restrict__ Gn, unsigned short* __restrict__ Tn,
                   int Npos, int total /* = B*Npos */) {
  const int wave = threadIdx.x >> 5, lane = threadIdx.x & 31;
  const int pos = blockIdx.x * 8 + wave;
  if (pos >= total) return;
  const int b = pos / Npos, n = pos - b * Npos;
  const float* gb = gen + (size_t)b * CC * Npos + n;
  const float* tb = tar + (size_t)b * CC * Npos + n;

  constexpr int NC = CC / 32;
  float gv[NC], tv[NC];
  float st = 0.f;
#pragma unroll
  for (int i = 0; i < NC; ++i) {
    const int c = lane + (i << 5);
    gv[i] = gb[(size_t)c * Npos];
    tv[i] = tb[(size_t)c * Npos];
    st += tv[i];
  }
  const float mean = waveAllSum(st) / (float)CC;
  float sg = 0.f, stt = 0.f;
#pragma unroll
  for (int i = 0; i < NC; ++i) {
    const float g = gv[i] - mean, t = tv[i] - mean;
    gv[i] = g; tv[i] = t;
    sg += g * g; stt += t * t;
  }
  sg = waveAllSum(sg);
  stt = waveAllSum(stt);
  const float rg = 1.f / (sqrtf(sg) + 1e-5f);
  const float rt = 1.f / (sqrtf(stt) + 1e-5f);
  const size_t ob = ((size_t)b * Npos + n) * CC;
#pragma unroll
  for (int i = 0; i < NC; ++i) {
    const int c = lane + (i << 5);
    Gn[ob + c] = f2bf(gv[i] * rg);
    Tn[ob + c] = f2bf(tv[i] * rt);
  }
}

// ---------------------------------------------------------------------------
// Kernel 2: fused bf16 WMMA GEMM  D[p,q] = Tn[b,p,:]·Gn[b,q,:],  d = (1-D)/2
// with one of three epilogues (no D materialization):
//   MODE 0: colminKey[q] = atomic-min over p of d
//   MODE 1: colsum[q]   += sum_p exp(INV_SIGMA - d*c1[q])
//   MODE 2: rowmax[p]    = atomic-max over q of exp(...)/(colsum[q]+1e-6)
// Block: 128x128 tile, 8 waves, wave tile 64(M)x32(N), BK=32, C compile-time
// so the whole async double-buffered pipeline is straight-line code.
// ---------------------------------------------------------------------------
#define LDST 40  // LDS row stride in halfs (80B, 16B-aligned)

template <int MODE, int C>
__global__ __launch_bounds__(256, 1)
void mrf_gemm(const unsigned short* __restrict__ Tn,
              const unsigned short* __restrict__ Gn,
              int Npos,
              unsigned* __restrict__ colminKey,
              const float* __restrict__ c1,
              float* __restrict__ colsum,
              unsigned* __restrict__ rowmaxBits) {
#if HAVE_ASYNC_LDS
  __shared__ unsigned short ldsA[2][128 * LDST];
  __shared__ unsigned short ldsB[2][128 * LDST];
#else
  __shared__ unsigned short ldsA[1][128 * LDST];
  __shared__ unsigned short ldsB[1][128 * LDST];
#endif

  const int b  = blockIdx.z;
  const int p0 = blockIdx.y * 128;
  const int q0 = blockIdx.x * 128;
  const int tid  = threadIdx.x;
  const int lane = tid & 31;
  const int wave = tid >> 5;
  const int wm   = wave & 1;   // 64-row half of M
  const int wn   = wave >> 1;  // 32-col quarter of N
  const int half = lane >> 4;  // lane-half selects K sub-range per ISA layout
  const int l16  = lane & 15;

  const size_t baseT = (size_t)b * Npos * C;
  const unsigned short* Ag = Tn + baseT + (size_t)p0 * C;
  const unsigned short* Bg = Gn + baseT + (size_t)q0 * C;

  v8f acc[4][2] = {};

  const int ldRow  = tid >> 1;
  const int ldSeg0 = tid & 1;

  union Frag { v16bf bf; uint4 q[2]; };

  // per-buffer compute step: fragments from LDS buffer `buf`, 8 WMMAs
  auto computeTile = [&](int buf) {
    Frag a[4], bfm[2];
#pragma unroll
    for (int mi = 0; mi < 4; ++mi) {
      const int arow = wm * 64 + mi * 16 + l16;
      a[mi].q[0] = *reinterpret_cast<const uint4*>(&ldsA[buf][arow * LDST + half * 8]);
      a[mi].q[1] = *reinterpret_cast<const uint4*>(&ldsA[buf][arow * LDST + 16 + half * 8]);
    }
#pragma unroll
    for (int ni = 0; ni < 2; ++ni) {
      const int brow = wn * 32 + ni * 16 + l16;
      bfm[ni].q[0] = *reinterpret_cast<const uint4*>(&ldsB[buf][brow * LDST + half * 8]);
      bfm[ni].q[1] = *reinterpret_cast<const uint4*>(&ldsB[buf][brow * LDST + 16 + half * 8]);
    }
#pragma unroll
    for (int mi = 0; mi < 4; ++mi)
#pragma unroll
      for (int ni = 0; ni < 2; ++ni)
        acc[mi][ni] = __builtin_amdgcn_wmma_f32_16x16x32_bf16(
            false, a[mi].bf, false, bfm[ni].bf, (short)0, acc[mi][ni], false, false);
  };

  constexpr int NK = C / 32;

#if HAVE_ASYNC_LDS
  // ---- async double-buffered pipeline (fully unrolled, branch-free) -------
  auto issueTile = [&](int buf, int kt) {
#pragma unroll
    for (int s = 0; s < 2; ++s) {
      const int seg = ldSeg0 + s * 2;  // 4 segments of 8 halfs per 32-half row
      const size_t goff = (size_t)ldRow * C + kt + seg * 8;
      GASYNC_B128(Ag + goff, &ldsA[buf][ldRow * LDST + seg * 8]);
      GASYNC_B128(Bg + goff, &ldsB[buf][ldRow * LDST + seg * 8]);
    }
  };
  issueTile(0, 0);
#pragma unroll
  for (int i = 0; i < NK - 1; ++i) {
    const int buf = i & 1;
    issueTile(buf ^ 1, (i + 1) * 32);  // prev-iter end barrier made this safe
    WAIT_ASYNC(4);                     // in-order: cur buffer's 4 ops landed
    __syncthreads();                   // all waves' tiles resident
    computeTile(buf);
    __syncthreads();                   // all reads done before buffer reuse
  }
  WAIT_ASYNC(0);
  __syncthreads();
  computeTile((NK - 1) & 1);           // epilogue is LDS-free: no end barrier
#else
  // ---- synchronous fallback ----------------------------------------------
#pragma unroll
  for (int i = 0; i < NK; ++i) {
    const int kt = i * 32;
#pragma unroll
    for (int s = 0; s < 2; ++s) {
      const int seg = ldSeg0 + s * 2;
      const size_t goff = (size_t)ldRow * C + kt + seg * 8;
      *reinterpret_cast<uint4*>(&ldsA[0][ldRow * LDST + seg * 8]) =
          *reinterpret_cast<const uint4*>(Ag + goff);
      *reinterpret_cast<uint4*>(&ldsB[0][ldRow * LDST + seg * 8]) =
          *reinterpret_cast<const uint4*>(Bg + goff);
    }
    __syncthreads();
    computeTile(0);
    __syncthreads();
  }
#endif

  // ----- epilogues -----
  // C/D layout: lane L holds column q=L%16, rows M = r + 8*(L/16) in v8f[r].
  if (MODE == 0) {  // column min of d
#pragma unroll
    for (int ni = 0; ni < 2; ++ni) {
      float cm = 3.4e38f;
#pragma unroll
      for (int mi = 0; mi < 4; ++mi)
#pragma unroll
        for (int r = 0; r < 8; ++r)
          cm = fminf(cm, 0.5f - 0.5f * acc[mi][ni][r]);
      cm = fminf(cm, __shfl_xor(cm, 16, 32));
      if (lane < 16) {
        const int q = q0 + wn * 32 + ni * 16 + l16;
        atomicMin(&colminKey[(size_t)b * Npos + q], fkeyEnc(cm));
      }
    }
  } else if (MODE == 1) {  // column sum of e
#pragma unroll
    for (int ni = 0; ni < 2; ++ni) {
      const int q = q0 + wn * 32 + ni * 16 + l16;
      const float c1v = c1[(size_t)b * Npos + q];
      float s = 0.f;
#pragma unroll
      for (int mi = 0; mi < 4; ++mi)
#pragma unroll
        for (int r = 0; r < 8; ++r)
          s += __expf(INV_SIGMA - (0.5f - 0.5f * acc[mi][ni][r]) * c1v);
      s += __shfl_xor(s, 16, 32);
      if (lane < 16)
        atomicAdd(&colsum[(size_t)b * Npos + q], s);
    }
  } else {  // MODE 2: row max of e/(colsum+1e-6)
    float c1v[2], rinv[2];
#pragma unroll
    for (int ni = 0; ni < 2; ++ni) {
      const int q = q0 + wn * 32 + ni * 16 + l16;
      c1v[ni]  = c1[(size_t)b * Npos + q];
      rinv[ni] = 1.f / (colsum[(size_t)b * Npos + q] + 1e-6f);
    }
#pragma unroll
    for (int mi = 0; mi < 4; ++mi) {
#pragma unroll
      for (int r = 0; r < 8; ++r) {
        float v = fmaxf(
            __expf(INV_SIGMA - (0.5f - 0.5f * acc[mi][0][r]) * c1v[0]) * rinv[0],
            __expf(INV_SIGMA - (0.5f - 0.5f * acc[mi][1][r]) * c1v[1]) * rinv[1]);
#pragma unroll
        for (int off = 8; off; off >>= 1) v = fmaxf(v, __shfl_xor(v, off, 32));
        if (l16 == 0) {  // lanes 0 and 16 hold row-maxes of their half-groups
          const int p = p0 + wm * 64 + mi * 16 + r + 8 * half;
          atomicMax(&rowmaxBits[(size_t)b * Npos + p], __float_as_uint(v));  // v>0
        }
      }
    }
  }
}

// ---------------------------------------------------------------------------
// Kernel 3: clamp column minima, precompute c1[q] = INV_SIGMA / colmin[q]
// ---------------------------------------------------------------------------
__global__ void mrf_colprep(const unsigned* __restrict__ key,
                            float* __restrict__ c1, int n) {
  const int i = blockIdx.x * blockDim.x + threadIdx.x;
  if (i >= n) return;
  float m = fkeyDec(key[i]);
  if (m < 1e-3f) m += 1e-3f;
  c1[i] = INV_SIGMA / m;
}

// ---------------------------------------------------------------------------
// Kernel 4: final reduction  loss = sum_b -log(mean_p rowmax)  (x1 for relu3,
// x2 for relu4 since style+content both use it)
// ---------------------------------------------------------------------------
__global__ __launch_bounds__(256)
void mrf_finalize(const unsigned* __restrict__ rm3, const unsigned* __restrict__ rm4,
                  float* __restrict__ out, int N3, int N4) {
  __shared__ float part[8];
  const int wave = threadIdx.x >> 5, lane = threadIdx.x & 31;
  const int lvl = wave >> 2;  // 0: relu3_2, 1: relu4_2
  const int b   = wave & 3;
  const unsigned* rm = lvl ? rm4 : rm3;
  const int N = lvl ? N4 : N3;
  float s = 0.f;
  for (int i = lane; i < N; i += 32) s += __uint_as_float(rm[(size_t)b * N + i]);
  s = waveAllSum(s);
  if (lane == 0) part[wave] = -logf(s / (float)N);
  __syncthreads();
  if (threadIdx.x == 0) {
    const float l3 = part[0] + part[1] + part[2] + part[3];
    const float l4 = part[4] + part[5] + part[6] + part[7];
    out[0] = l3 + 2.0f * l4;
  }
}

// ---------------------------------------------------------------------------
// Host launcher
// ---------------------------------------------------------------------------
extern "C" void kernel_launch(void* const* d_in, const int* in_sizes, int n_in,
                              void* d_out, int out_size, void* d_ws, size_t ws_size,
                              hipStream_t stream) {
  (void)in_sizes; (void)n_in; (void)out_size; (void)ws_size;
  const float* gen3 = (const float*)d_in[0];
  const float* tar3 = (const float*)d_in[1];
  const float* gen4 = (const float*)d_in[2];
  const float* tar4 = (const float*)d_in[3];
  constexpr int B = 4, C3 = 256, N3 = 64 * 64, C4 = 512, N4 = 32 * 32;

  char* ws = (char*)d_ws;
  size_t off = 0;
  auto alloc = [&](size_t bytes) -> void* {
    void* p = ws + off;
    off = (off + bytes + 255) & ~(size_t)255;
    return p;
  };
  unsigned short* Tn3 = (unsigned short*)alloc((size_t)B * N3 * C3 * 2);
  unsigned short* Gn3 = (unsigned short*)alloc((size_t)B * N3 * C3 * 2);
  unsigned short* Tn4 = (unsigned short*)alloc((size_t)B * N4 * C4 * 2);
  unsigned short* Gn4 = (unsigned short*)alloc((size_t)B * N4 * C4 * 2);
  unsigned* cmin3 = (unsigned*)alloc((size_t)B * N3 * 4);
  float*    c1_3  = (float*)   alloc((size_t)B * N3 * 4);
  float*    csum3 = (float*)   alloc((size_t)B * N3 * 4);
  unsigned* rmax3 = (unsigned*)alloc((size_t)B * N3 * 4);
  unsigned* cmin4 = (unsigned*)alloc((size_t)B * N4 * 4);
  float*    c1_4  = (float*)   alloc((size_t)B * N4 * 4);
  float*    csum4 = (float*)   alloc((size_t)B * N4 * 4);
  unsigned* rmax4 = (unsigned*)alloc((size_t)B * N4 * 4);

  // init stat buffers (graph-capture safe async memsets)
  (void)hipMemsetAsync(cmin3, 0xFF, (size_t)B * N3 * 4, stream);
  (void)hipMemsetAsync(cmin4, 0xFF, (size_t)B * N4 * 4, stream);
  (void)hipMemsetAsync(csum3, 0,    (size_t)B * N3 * 4, stream);
  (void)hipMemsetAsync(csum4, 0,    (size_t)B * N4 * 4, stream);
  (void)hipMemsetAsync(rmax3, 0,    (size_t)B * N3 * 4, stream);
  (void)hipMemsetAsync(rmax4, 0,    (size_t)B * N4 * 4, stream);

  // normalize (one wave per spatial position)
  mrf_normalize<C3><<<(B * N3) / 8, 256, 0, stream>>>(gen3, tar3, Gn3, Tn3, N3, B * N3);
  mrf_normalize<C4><<<(B * N4) / 8, 256, 0, stream>>>(gen4, tar4, Gn4, Tn4, N4, B * N4);

  const dim3 grid3(N3 / 128, N3 / 128, B), grid4(N4 / 128, N4 / 128, B);

  // pass 1: column minima
  mrf_gemm<0, C3><<<grid3, 256, 0, stream>>>(Tn3, Gn3, N3, cmin3, c1_3, csum3, rmax3);
  mrf_gemm<0, C4><<<grid4, 256, 0, stream>>>(Tn4, Gn4, N4, cmin4, c1_4, csum4, rmax4);
  mrf_colprep<<<(B * N3 + 255) / 256, 256, 0, stream>>>(cmin3, c1_3, B * N3);
  mrf_colprep<<<(B * N4 + 255) / 256, 256, 0, stream>>>(cmin4, c1_4, B * N4);

  // pass 2: column exp-sums
  mrf_gemm<1, C3><<<grid3, 256, 0, stream>>>(Tn3, Gn3, N3, cmin3, c1_3, csum3, rmax3);
  mrf_gemm<1, C4><<<grid4, 256, 0, stream>>>(Tn4, Gn4, N4, cmin4, c1_4, csum4, rmax4);

  // pass 3: row maxima of normalized cs
  mrf_gemm<2, C3><<<grid3, 256, 0, stream>>>(Tn3, Gn3, N3, cmin3, c1_3, csum3, rmax3);
  mrf_gemm<2, C4><<<grid4, 256, 0, stream>>>(Tn4, Gn4, N4, cmin4, c1_4, csum4, rmax4);

  // final scalar
  mrf_finalize<<<1, 256, 0, stream>>>(rmax3, rmax4, (float*)d_out, N3, N4);
}
